// ExpertLinear_11948599017885
// MI455X (gfx1250) — matrix operand order repaired and checked by reading, not compiled, not run on hardware
//
#include <hip/hip_runtime.h>

// Blended MoE linear as ONE f16 WMMA GEMM:
//   y[b,o] = sum_{k,i} (ew[b,k]*x[b,i]) * W[k,o,i]  ->  [8192 x 8192] @ [8192 x 1024]
// A' generated on the fly (scale by routing weight during fp32->f16 convert).
// Register-staged global prefetch + LDS double buffering: one barrier per K=64 tile,
// 16 v_wmma_f32_16x16x32_f16 per iteration. All global indexing is 32-bit
// (buffers < 2^31 bytes) so addresses stay SGPR-base + 32-bit VGPR offset.

typedef __attribute__((ext_vector_type(16))) _Float16 v16h;
typedef __attribute__((ext_vector_type(8)))  _Float16 v8h;
typedef __attribute__((ext_vector_type(4)))  _Float16 v4h;
typedef __attribute__((ext_vector_type(8)))  float    v8f;

#define B_ROWS   8192
#define IN_DIM   1024
#define OUT_DIM  1024
#define NEXP     8
#define KTOT     (NEXP * IN_DIM)      // 8192 reduction length

#define BM 128
#define BN 128
#define BK 64                          // two WMMA k-steps per LDS tile
#define NKB (KTOT / BK)                // 128 iterations
#define LDK 72                         // padded LDS row stride in halves (144B, 16B-aligned)
#define THREADS 256                    // 8 waves (wave32)
// wave grid 2 (M) x 4 (N); wave tile 64x32 -> 4x2 subtiles of 16x16

__global__ __launch_bounds__(THREADS)
void moe_blend_gemm(const float* __restrict__ x,
                    const float* __restrict__ ew,
                    const float* __restrict__ w,
                    const float* __restrict__ bias,
                    float* __restrict__ out)
{
    __shared__ __align__(16) _Float16 As[2][BM * LDK];  // A' tiles: [m][k] f16, double buffered
    __shared__ __align__(16) _Float16 Bs[2][BN * LDK];  // W  tiles: [n][k] f16, double buffered
    __shared__ float EWs[BM * NEXP];                    // routing weights for row block

    const uint tid = threadIdx.x;
    const uint m0  = blockIdx.y * BM;
    const uint n0  = blockIdx.x * BN;

    const uint wave   = tid >> 5;
    const uint lane   = tid & 31;
    const uint half16 = lane >> 4;     // 0|1
    const uint l16    = lane & 15;
    const uint wm     = wave >> 2;     // 0..1
    const uint wn     = wave & 3;      // 0..3

    // Per-thread loop-invariant address components (32-bit element offsets).
    uint xoff[8], woff[8], lrow[8], lcol[8];
    #pragma unroll
    for (uint j = 0; j < 8; ++j) {
        const uint linear = tid + THREADS * j;   // 0..2047
        const uint row    = linear >> 4;         // 0..127
        const uint colk   = (linear & 15) * 4;   // 0..60
        lrow[j] = row;
        lcol[j] = colk;
        xoff[j] = (m0 + row) * IN_DIM + colk;    // x element offset (row part)
        woff[j] = (n0 + row) * IN_DIM + colk;    // w element offset (row part, no expert)
    }

    // ---- global -> register staging (8 float4 per tile per thread) ----
    float4 ra[8], rb[8];
    auto load_tile = [&](uint kb) {
        const uint kk0   = kb * BK;
        const uint e     = kk0 >> 10;                        // expert (BK | 1024: no straddle)
        const uint i0    = kk0 & (IN_DIM - 1);
        const uint wbase = e * (OUT_DIM * IN_DIM) + i0;      // uniform scalar per iteration
        #pragma unroll
        for (uint j = 0; j < 8; ++j) {
            ra[j] = *reinterpret_cast<const float4*>(x + (xoff[j] + i0));
            rb[j] = *reinterpret_cast<const float4*>(w + (woff[j] + wbase));
        }
    };

    // ---- register -> LDS (fp32 -> f16 convert, A scaled by routing weight) ----
    auto store_tile = [&](uint kb, _Float16* __restrict__ Ab, _Float16* __restrict__ Bb) {
        const uint e = (kb * BK) >> 10;
        #pragma unroll
        for (uint j = 0; j < 8; ++j) {
            const float s = EWs[lrow[j] * NEXP + e];
            v4h ah = { (_Float16)(ra[j].x * s), (_Float16)(ra[j].y * s),
                       (_Float16)(ra[j].z * s), (_Float16)(ra[j].w * s) };
            *reinterpret_cast<v4h*>(&Ab[lrow[j] * LDK + lcol[j]]) = ah;
            v4h bh = { (_Float16)rb[j].x, (_Float16)rb[j].y,
                       (_Float16)rb[j].z, (_Float16)rb[j].w };
            *reinterpret_cast<v4h*>(&Bb[lrow[j] * LDK + lcol[j]]) = bh;
        }
    };

    v8f acc[4][2] = {};

    // ---- LDS -> fragments -> 16 WMMAs (2 k-steps of K=32) ----
    auto compute_tile = [&](const _Float16* __restrict__ Ab,
                            const _Float16* __restrict__ Bb) {
        #pragma unroll
        for (uint ks = 0; ks < 2; ++ks) {
            const uint kb32 = ks * 32;
            // A frag (16x32): elems 0..15 = K 16h..16h+15 contiguous
            v16h afrag[4];
            #pragma unroll
            for (uint mt = 0; mt < 4; ++mt) {
                const uint r = wm * 64 + mt * 16 + l16;
                v8h lo = *reinterpret_cast<const v8h*>(&Ab[r * LDK + kb32 + 16 * half16]);
                v8h hi = *reinterpret_cast<const v8h*>(&Ab[r * LDK + kb32 + 16 * half16 + 8]);
                afrag[mt] = __builtin_shufflevector(lo, hi,
                    0,1,2,3,4,5,6,7,8,9,10,11,12,13,14,15);
            }
            // B frag (32x16): lane = col n, elems 0..15 = K 16h..16h+15 contiguous
            v16h bfrag[2];
            #pragma unroll
            for (uint nt = 0; nt < 2; ++nt) {
                const uint rn = wn * 32 + nt * 16 + l16;
                v8h lo = *reinterpret_cast<const v8h*>(&Bb[rn * LDK + kb32 + 16 * half16]);
                v8h hi = *reinterpret_cast<const v8h*>(&Bb[rn * LDK + kb32 + 16 * half16 + 8]);
                bfrag[nt] = __builtin_shufflevector(lo, hi,
                    0,1,2,3,4,5,6,7,8,9,10,11,12,13,14,15);
            }
            #pragma unroll
            for (uint mt = 0; mt < 4; ++mt)
                #pragma unroll
                for (uint nt = 0; nt < 2; ++nt)
                    acc[mt][nt] = __builtin_amdgcn_wmma_f32_16x16x32_f16(
                        false, afrag[mt], false, bfrag[nt],
                        (short)0, acc[mt][nt], false, false);
        }
    };

    // ---- prologue: EW preload overlaps tile-0 global loads ----
    load_tile(0);
    for (uint idx = tid; idx < BM * NEXP; idx += THREADS)
        EWs[idx] = ew[(m0 + (idx >> 3)) * NEXP + (idx & 7)];
    __syncthreads();                 // EWs visible (store_tile reads it)
    store_tile(0, As[0], Bs[0]);
    __syncthreads();                 // buffer 0 visible

    // ---- main loop: prefetch k+1 under compute of k; one barrier per tile ----
    uint cur = 0;
    #pragma unroll 1
    for (uint kb = 0; kb + 1 < NKB; ++kb) {
        load_tile(kb + 1);                       // global loads issue, hide under WMMA
        compute_tile(As[cur], Bs[cur]);
        store_tile(kb + 1, As[cur ^ 1], Bs[cur ^ 1]);
        __syncthreads();
        cur ^= 1;
    }
    compute_tile(As[cur], Bs[cur]);              // last tile

    // ---- epilogue: add blended bias (ew @ bias) and store ----
    // C/D layout: vgpr r, lane(h,l16) -> M = r + 8h, N = l16.
    #pragma unroll
    for (uint nt = 0; nt < 2; ++nt) {
        const uint n = n0 + wn * 32 + nt * 16 + l16;
        float bv[NEXP];
        #pragma unroll
        for (uint k = 0; k < NEXP; ++k) bv[k] = bias[k * OUT_DIM + n];

        #pragma unroll
        for (uint mt = 0; mt < 4; ++mt) {
            #pragma unroll
            for (uint r = 0; r < 8; ++r) {
                const uint ml = wm * 64 + mt * 16 + r + 8 * half16;
                float bs = 0.0f;
                #pragma unroll
                for (uint k = 0; k < NEXP; ++k) bs += EWs[ml * NEXP + k] * bv[k];
                out[(m0 + ml) * OUT_DIM + n] = acc[mt][nt][r] + bs;
            }
        }
    }
}

extern "C" void kernel_launch(void* const* d_in, const int* in_sizes, int n_in,
                              void* d_out, int out_size, void* d_ws, size_t ws_size,
                              hipStream_t stream) {
    (void)in_sizes; (void)n_in; (void)out_size; (void)d_ws; (void)ws_size;
    const float* x    = (const float*)d_in[0];  // [8192, 1024]
    const float* ew   = (const float*)d_in[1];  // [8192, 8]
    const float* w    = (const float*)d_in[2];  // [8, 1024, 1024]
    const float* bias = (const float*)d_in[3];  // [8, 1024]
    float* out        = (float*)d_out;          // [8192, 1024]

    dim3 grid(OUT_DIM / BN, B_ROWS / BM);       // (8, 64)
    dim3 block(THREADS);
    hipLaunchKernelGGL(moe_blend_gemm, grid, block, 0, stream, x, ew, w, bias, out);
}
